// LocalCrossAttention_9904194584598
// MI455X (gfx1250) — matrix operand order repaired and checked by reading, not compiled, not run on hardware
//
#include <hip/hip_runtime.h>
#include <math.h>

// ---------------------------------------------------------------------------
// CDNA5 (gfx1250) fused bidirectional sigmoid cross-attention.
// Pipeline:
//   1) transpose_to_bf16: X [.,C,L] f32 -> XT [.,L,C] bf16 via async-to-LDS
//      staging (also used for the weight matrices, B=1).
//   2) proj_bf16: Y = XT @ W + b, pure v_wmma_f32_16x16x32_bf16 with b128
//      fragment loads (V stored transposed for the ctx matmul).
//   3) sigmoid_attn: flash-style probs=sigmoid(QK^T*s), ctx=probs@V fused,
//      P exchanged through LDS as bf16.
// ---------------------------------------------------------------------------

typedef __attribute__((ext_vector_type(16))) __bf16       v16bf;
typedef __attribute__((ext_vector_type(8)))  float        v8f;
typedef __attribute__((ext_vector_type(4)))  unsigned int u32x4;

union FragAB {
    v16bf          v;
    u32x4          q[2];
    unsigned short u16[16];
};

__device__ __forceinline__ unsigned short f2bf(float f) {
    unsigned int x = __float_as_uint(f);
    x += 0x7FFFu + ((x >> 16) & 1u);            // round-to-nearest-even
    return (unsigned short)(x >> 16);
}

// ---------------------------------------------------------------------------
// Transpose + convert: dst[b][l][c] (bf16) = src[b][c][l] (f32).
// Tile 64(l) x 32(c), staged through LDS with async global->LDS b128 copies.
// grid = (L/64, C/32, B), block = 256.
// ---------------------------------------------------------------------------
#define TP_STRIDE 68   // padded f32 row stride (dwords) -> low LDS bank conflict

__global__ void transpose_to_bf16_kernel(const float* __restrict__ src,
                                         unsigned short* __restrict__ dst,
                                         int C, int L)
{
    __shared__ __align__(16) float T[32 * TP_STRIDE];

    const int t  = threadIdx.x;
    const int l0 = blockIdx.x << 6;
    const int c0 = blockIdx.y << 5;
    const int b  = blockIdx.z;
    const float* sb = src + (size_t)b * C * L;

    // ---- async-stage tile into LDS as [c][l] f32, 16B per lane per issue ----
    #pragma unroll
    for (int i = 0; i < 2; ++i) {
        const int idx = (i << 8) + t;          // 0..511 chunks of 4 floats
        const int c   = idx >> 4;              // 0..31
        const int l4  = (idx & 15) << 2;       // 0,4,..,60
        const float* gp = sb + (size_t)(c0 + c) * L + l0 + l4;
        const unsigned lds_off = (unsigned)(size_t)(&T[c * TP_STRIDE + l4]);
        asm volatile("global_load_async_to_lds_b128 %0, %1, off"
                     :: "v"(lds_off), "v"(gp)
                     : "memory");
    }
    asm volatile("s_wait_asynccnt 0x0" ::: "memory");
    __syncthreads();

    // ---- write out transposed bf16, 4 channels (b64) per store ----
    #pragma unroll
    for (int i = 0; i < 2; ++i) {
        const int l  = (i << 5) + (t >> 3);    // 0..63
        const int c4 = (t & 7) << 2;           // 0,4,..,28
        unsigned long long p = 0;
        #pragma unroll
        for (int j = 0; j < 4; ++j)
            p |= (unsigned long long)f2bf(T[(c4 + j) * TP_STRIDE + l]) << (16 * j);
        *(unsigned long long*)(dst + (size_t)b * L * C + (size_t)(l0 + l) * C
                                   + c0 + c4) = p;
    }
}

// ---------------------------------------------------------------------------
// Projection GEMM: Y[row,n] = sum_c XT[row,c]*WT[n,c] + bias[n]  (bf16 out).
// Wave tile 16 rows x 64 cols; 4 waves/block -> 64x64 block tile.
// All fragments are direct b128 loads from row-major-along-K bf16 operands.
// transposed==1 stores per-batch Y^T as [B,512,L] (V operand for ctx matmul).
// grid = (512/64, B*L/64), block = 128.
// ---------------------------------------------------------------------------
__global__ void proj_bf16_kernel(const unsigned short* __restrict__ XT, // [B*L, C]
                                 const unsigned short* __restrict__ WT, // [512, C]
                                 const float* __restrict__ bias,        // [512]
                                 unsigned short* __restrict__ Y,
                                 int C, int L, int transposed)
{
    const int wave = threadIdx.x >> 5;             // 0..3
    const int lane = threadIdx.x & 31;
    const int hf   = lane >> 4;
    const int mn   = lane & 15;
    const int nbase   = blockIdx.x << 6;
    const int rowbase = (blockIdx.y << 6) + (wave << 4);

    const unsigned short* ap = XT + (size_t)(rowbase + mn) * C + (hf << 3);

    v8f acc[4];
    #pragma unroll
    for (int i = 0; i < 4; ++i) acc[i] = (v8f)0.0f;

    for (int c0 = 0; c0 < C; c0 += 32) {
        FragAB a;
        a.q[0] = *(const u32x4*)(ap + c0);         // A: K hf*8..+7
        a.q[1] = *(const u32x4*)(ap + c0 + 16);    // A: K 16+hf*8..+7
        #pragma unroll
        for (int sub = 0; sub < 4; ++sub) {
            const unsigned short* bp = WT + (size_t)(nbase + (sub << 4) + mn) * C
                                          + c0 + (hf << 3);
            FragAB bb;
            bb.q[0] = *(const u32x4*)(bp);         // B: K hf*8..+7
            bb.q[1] = *(const u32x4*)(bp + 16);    // B: K 16+hf*8..+7
            acc[sub] = __builtin_amdgcn_wmma_f32_16x16x32_bf16(false, a.v, false, bb.v,
                                                               (short)0, acc[sub],
                                                               false, false);
        }
    }

    const int b  = rowbase / L;
    const int l0 = rowbase % L;
    #pragma unroll
    for (int sub = 0; sub < 4; ++sub) {
        const int   n  = nbase + (sub << 4) + mn;
        const float bv = bias[n];
        if (!transposed) {
            unsigned short* op = Y + (size_t)(rowbase + (hf << 3)) * 512 + n;
            #pragma unroll
            for (int r = 0; r < 8; ++r)
                op[(size_t)r * 512] = f2bf(acc[sub][r] + bv);
        } else {
            unsigned short* op = Y + (size_t)b * 512 * L + (size_t)n * L
                                   + l0 + (hf << 3);
            #pragma unroll
            for (int r = 0; r < 8; ++r)
                op[r] = f2bf(acc[sub][r] + bv);
        }
    }
}

// ---------------------------------------------------------------------------
// Fused sigmoid attention (unchanged from round 1):
//   probs = sigmoid(Q K^T * scale) (f32 out), ctx = probs @ V.
// Block = 256 threads (8 waves) -> 32 queries x 512 ctx cols; key blocks of 64.
// ---------------------------------------------------------------------------
__global__ void sigmoid_attn_kernel(const unsigned short* __restrict__ Q,   // [B,Lq,512]
                                    const unsigned short* __restrict__ K,   // [B,Lk,512]
                                    const unsigned short* __restrict__ VT,  // [B,512,Lk]
                                    float* __restrict__ ctx,                // [B,Lq,512]
                                    float* __restrict__ probs,              // [B,Lq,Lk]
                                    int Lq, int Lk)
{
    __shared__ __align__(16) unsigned short Plds[32 * 64];   // 4 KB bf16 P tile

    const float scale = 0.044194173824159216f;               // 1/sqrt(512)

    const int wave = threadIdx.x >> 5;   // 0..7
    const int lane = threadIdx.x & 31;
    const int hf   = lane >> 4;
    const int mn   = lane & 15;
    const int wm   = wave & 1;
    const int wn   = wave >> 1;          // 0..3
    const int b    = blockIdx.y;
    const int m0   = blockIdx.x << 5;

    const unsigned short* Qb  = Q  + (size_t)b * Lq * 512;
    const unsigned short* Kb  = K  + (size_t)b * Lk * 512;
    const unsigned short* VTb = VT + (size_t)b * 512 * Lk;
    float* ctxb   = ctx   + (size_t)b * Lq * 512;
    float* probsb = probs + (size_t)b * Lq * Lk;

    v8f acc[8];
    #pragma unroll
    for (int i = 0; i < 8; ++i) acc[i] = (v8f)0.0f;

    const unsigned short* qp = Qb + (size_t)(m0 + (wm << 4) + mn) * 512 + (hf << 3);

    for (int kblk = 0; kblk < Lk; kblk += 64) {
        // ---- S tile (16x16 per wave) ----
        v8f s = (v8f)0.0f;
        const unsigned short* kp = Kb + (size_t)(kblk + (wn << 4) + mn) * 512 + (hf << 3);
        #pragma unroll 4
        for (int e0 = 0; e0 < 512; e0 += 32) {
            FragAB a, bb;
            a.q[0]  = *(const u32x4*)(qp + e0);
            a.q[1]  = *(const u32x4*)(qp + e0 + 16);
            bb.q[0] = *(const u32x4*)(kp + e0);
            bb.q[1] = *(const u32x4*)(kp + e0 + 16);
            s = __builtin_amdgcn_wmma_f32_16x16x32_bf16(false, a.v, false, bb.v,
                                                        (short)0, s, false, false);
        }

        __syncthreads();
        #pragma unroll
        for (int r = 0; r < 8; ++r) {
            const float p  = 1.0f / (1.0f + __expf(-s[r] * scale));
            const int   mm = (wm << 4) + (hf << 3) + r;
            probsb[(size_t)(m0 + mm) * Lk + kblk + (wn << 4) + mn] = p;
            Plds[mm * 64 + (wn << 4) + mn] = f2bf(p);
        }
        __syncthreads();

        // ---- ctx(16x128 per wave) += P(16x64) @ Vblk(64x128) ----
        #pragma unroll
        for (int sub = 0; sub < 8; ++sub) {
            const unsigned short* vp = VTb + (size_t)((wn << 7) + (sub << 4) + mn) * Lk
                                         + kblk + (hf << 3);
            #pragma unroll
            for (int ks = 0; ks < 2; ++ks) {
                FragAB a, bb;
                const unsigned short* pp =
                    &Plds[((wm << 4) + mn) * 64 + (ks << 5) + (hf << 3)];
                a.q[0]  = *(const u32x4*)(pp);
                a.q[1]  = *(const u32x4*)(pp + 16);
                bb.q[0] = *(const u32x4*)(vp + (ks << 5));
                bb.q[1] = *(const u32x4*)(vp + (ks << 5) + 16);
                acc[sub] = __builtin_amdgcn_wmma_f32_16x16x32_bf16(false, a.v, false, bb.v,
                                                                   (short)0, acc[sub],
                                                                   false, false);
            }
        }
    }

    #pragma unroll
    for (int sub = 0; sub < 8; ++sub) {
        float* op = ctxb + (size_t)(m0 + (wm << 4) + (hf << 3)) * 512
                         + (wn << 7) + (sub << 4) + mn;
        #pragma unroll
        for (int r = 0; r < 8; ++r)
            op[(size_t)r * 512] = acc[sub][r];
    }
}

// ---------------------------------------------------------------------------
// Host launch
// ---------------------------------------------------------------------------
extern "C" void kernel_launch(void* const* d_in, const int* in_sizes, int n_in,
                              void* d_out, int out_size, void* d_ws, size_t ws_size,
                              hipStream_t stream) {
    const int B = 16, L1 = 2048, L2 = 1024, E = 512, C2 = 768;

    const float* in1 = (const float*)d_in[0];   // [B,512,L1]
    const float* in2 = (const float*)d_in[1];   // [B,768,L2]
    const float* Wq1 = (const float*)d_in[2];
    const float* bq1 = (const float*)d_in[3];
    const float* Wk1 = (const float*)d_in[4];
    const float* bk1 = (const float*)d_in[5];
    const float* Wv1 = (const float*)d_in[6];
    const float* bv1 = (const float*)d_in[7];
    const float* Wq2 = (const float*)d_in[8];
    const float* bq2 = (const float*)d_in[9];
    const float* Wk2 = (const float*)d_in[10];
    const float* bk2 = (const float*)d_in[11];
    const float* Wv2 = (const float*)d_in[12];
    const float* bv2 = (const float*)d_in[13];

    // bf16 workspace layout (element counts)
    const size_t n1  = (size_t)B * L1 * E;      // q1/k1/v1T
    const size_t n2  = (size_t)B * L2 * E;      // q2/k2/v2T
    const size_t nx1 = (size_t)B * L1 * E;      // XT1 [B,L1,512]
    const size_t nx2 = (size_t)B * L2 * C2;     // XT2 [B,L2,768]
    const size_t nw1 = (size_t)E * E;           // WT (512x512)
    const size_t nw2 = (size_t)E * C2;          // WT (512x768)

    unsigned short* q1   = (unsigned short*)d_ws;
    unsigned short* k1   = q1   + n1;
    unsigned short* v1T  = k1   + n1;
    unsigned short* q2   = v1T  + n1;
    unsigned short* k2   = q2   + n2;
    unsigned short* v2T  = k2   + n2;
    unsigned short* XT1  = v2T  + n2;
    unsigned short* XT2  = XT1  + nx1;
    unsigned short* WTq1 = XT2  + nx2;
    unsigned short* WTk1 = WTq1 + nw1;
    unsigned short* WTv1 = WTk1 + nw1;
    unsigned short* WTq2 = WTv1 + nw1;
    unsigned short* WTk2 = WTq2 + nw2;
    unsigned short* WTv2 = WTk2 + nw2;

    // outputs, concatenated: ctx2, probs2, ctx1, probs1
    float* out    = (float*)d_out;
    float* ctx2   = out;
    float* probs2 = ctx2   + (size_t)B * L1 * E;
    float* ctx1   = probs2 + (size_t)B * L1 * L2;
    float* probs1 = ctx1   + (size_t)B * L2 * E;

    // ---- stage 1: transpose+convert activations and weights to bf16 ----
    const dim3 tblk(256);
    transpose_to_bf16_kernel<<<dim3(L1 / 64, E / 32, B),  tblk, 0, stream>>>(in1, XT1, E,  L1);
    transpose_to_bf16_kernel<<<dim3(L2 / 64, C2 / 32, B), tblk, 0, stream>>>(in2, XT2, C2, L2);
    transpose_to_bf16_kernel<<<dim3(E / 64, E / 32, 1),   tblk, 0, stream>>>(Wq1, WTq1, E,  E);
    transpose_to_bf16_kernel<<<dim3(E / 64, E / 32, 1),   tblk, 0, stream>>>(Wk1, WTk1, E,  E);
    transpose_to_bf16_kernel<<<dim3(E / 64, E / 32, 1),   tblk, 0, stream>>>(Wv1, WTv1, E,  E);
    transpose_to_bf16_kernel<<<dim3(E / 64, C2 / 32, 1),  tblk, 0, stream>>>(Wq2, WTq2, C2, E);
    transpose_to_bf16_kernel<<<dim3(E / 64, C2 / 32, 1),  tblk, 0, stream>>>(Wk2, WTk2, C2, E);
    transpose_to_bf16_kernel<<<dim3(E / 64, C2 / 32, 1),  tblk, 0, stream>>>(Wv2, WTv2, C2, E);

    // ---- stage 2: six WMMA projection GEMMs ----
    const dim3 pblk(128);
    const dim3 pg1(E / 64, (B * L1) / 64);
    const dim3 pg2(E / 64, (B * L2) / 64);
    proj_bf16_kernel<<<pg1, pblk, 0, stream>>>(XT1, WTq1, bq1, q1,  E,  L1, 0);
    proj_bf16_kernel<<<pg1, pblk, 0, stream>>>(XT1, WTk1, bk1, k1,  E,  L1, 0);
    proj_bf16_kernel<<<pg1, pblk, 0, stream>>>(XT1, WTv1, bv1, v1T, E,  L1, 1);
    proj_bf16_kernel<<<pg2, pblk, 0, stream>>>(XT2, WTq2, bq2, q2,  C2, L2, 0);
    proj_bf16_kernel<<<pg2, pblk, 0, stream>>>(XT2, WTk2, bk2, k2,  C2, L2, 0);
    proj_bf16_kernel<<<pg2, pblk, 0, stream>>>(XT2, WTv2, bv2, v2T, C2, L2, 1);

    // ---- stage 3: fused sigmoid attention, both directions ----
    const dim3 ablk(256);
    sigmoid_attn_kernel<<<dim3(L1 / 32, B), ablk, 0, stream>>>(q1, k2, v2T,
                                                               ctx2, probs2, L1, L2);
    sigmoid_attn_kernel<<<dim3(L2 / 32, B), ablk, 0, stream>>>(q2, k1, v1T,
                                                               ctx1, probs1, L2, L1);
}
